// PersistentObserver_61761629716835
// MI455X (gfx1250) — compile-verified, matrix operand order checked by reading, not run on hardware
//
#include <hip/hip_runtime.h>
#include <math.h>

#define BATCH 4
#define NNODE 256
#define OBSD  64
#define QDIMI 32
#define HID   256
#define LATD  128
#define NOPSD 8
#define QENCD 64

// output layout: edge_logits [4*256*256], op_logits [4*256*256*8], next_pred [4*256]
#define EDGE_OFF 0
#define OP_OFF   262144
#define NEXT_OFF 2359296

typedef __attribute__((ext_vector_type(8)))  float          v8f;
typedef __attribute__((ext_vector_type(16))) __bf16         v16bf;
typedef __attribute__((ext_vector_type(4)))  unsigned int   v4i;
typedef __attribute__((ext_vector_type(8)))  unsigned int   v8i;

__device__ __forceinline__ unsigned short f32_to_bf16(float f) {
  union { float f; unsigned int u; } v; v.f = f;
  unsigned int u = v.u + 0x7FFFu + ((v.u >> 16) & 1u);   // round-to-nearest-even
  return (unsigned short)(u >> 16);
}

// packed |a - b| on two bf16 lanes per dword: v_pk_add_bf16 with src1 negated, then clear signs
__device__ __forceinline__ unsigned int pk_absdiff_bf16(unsigned int a, unsigned int b) {
  unsigned int d;
  asm("v_pk_add_bf16 %0, %1, %2 neg_lo:[0,1] neg_hi:[0,1]" : "=v"(d) : "v"(a), "v"(b));
  return d & 0x7FFF7FFFu;
}

#define WMMA_BF16(A, B, C) \
  __builtin_amdgcn_wmma_f32_16x16x32_bf16(false, (A), false, (B), (short)0, (C), false, false)

// ---------------- per-node (tiny) kernels: exact f32 VALU ----------------

__global__ void k_enc1(const float* __restrict__ x, const float* __restrict__ w,
                       const float* __restrict__ b, float* __restrict__ y) {
  int tid = blockIdx.x * 256 + threadIdx.x;           // 1024*256
  int row = tid >> 8, c = tid & 255;
  const float* xr = x + row * OBSD;
  float s = b[c];
  for (int k = 0; k < OBSD; ++k) s += xr[k] * w[k * HID + c];
  y[tid] = fmaxf(s, 0.0f);
}

__global__ void k_enc2(const float* __restrict__ x, const float* __restrict__ w,
                       const float* __restrict__ b, float* __restrict__ y) {
  int tid = blockIdx.x * 256 + threadIdx.x;           // 1024*256
  int row = tid >> 8, c = tid & 255;
  const float* xr = x + row * HID;
  float s = b[c];
  for (int k = 0; k < HID; ++k) s += xr[k] * w[k * HID + c];
  y[tid] = fmaxf(s, 0.0f);
}

__global__ void k_upd(const float* __restrict__ emb, const float* __restrict__ w,
                      const float* __restrict__ b, float* __restrict__ y) {
  int tid = blockIdx.x * 256 + threadIdx.x;           // 1024*128
  int row = tid >> 7, c = tid & 127;
  const float* xr = emb + row * HID;
  float s = b[c];
  for (int k = 0; k < HID; ++k) s += xr[k] * w[k * LATD + c];
  y[tid] = s;
}

__global__ void k_gru(const float* __restrict__ upd, const float* __restrict__ latent,
                      const float* __restrict__ wi, const float* __restrict__ bi,
                      const float* __restrict__ wh, const float* __restrict__ bh,
                      float* __restrict__ latf, unsigned short* __restrict__ latb) {
  int tid = blockIdx.x * 256 + threadIdx.x;           // 1024*128
  int row = tid >> 7, c = tid & 127;
  const float* u = upd + row * LATD;
  const float* h = latent + row * LATD;
  float ir = bi[c], iz = bi[c + 128], inn = bi[c + 256];
  float hr = bh[c], hz = bh[c + 128], hn  = bh[c + 256];
  for (int k = 0; k < LATD; ++k) {
    float uk = u[k]; const float* wr = wi + k * 384;
    ir += uk * wr[c]; iz += uk * wr[c + 128]; inn += uk * wr[c + 256];
    float hk = h[k]; const float* wr2 = wh + k * 384;
    hr += hk * wr2[c]; hz += hk * wr2[c + 128]; hn += hk * wr2[c + 256];
  }
  float r = 1.0f / (1.0f + expf(-(ir + hr)));
  float z = 1.0f / (1.0f + expf(-(iz + hz)));
  float nh = tanhf(inn + r * hn);
  float lv = (1.0f - z) * nh + z * h[c];
  latf[tid] = lv;
  latb[tid] = f32_to_bf16(lv);
}

__global__ void k_qemb(const float* __restrict__ q, const float* __restrict__ w,
                       const float* __restrict__ b, float* __restrict__ y) {
  int tid = blockIdx.x * 256 + threadIdx.x;           // 1024*64
  int row = tid >> 6, c = tid & 63;
  const float* xr = q + row * QDIMI;
  float s = b[c];
  for (int k = 0; k < QDIMI; ++k) s += xr[k] * w[k * QENCD + c];
  y[tid] = fmaxf(s, 0.0f);
}

__global__ void k_nh1(const float* __restrict__ latf, const float* __restrict__ emb,
                      const float* __restrict__ qe, const float* __restrict__ w,
                      const float* __restrict__ b, float* __restrict__ y) {
  int tid = blockIdx.x * 256 + threadIdx.x;           // 1024*256
  int row = tid >> 8, c = tid & 255;
  float s = b[c];
  const float* l = latf + row * LATD;
  for (int k = 0; k < LATD; ++k)  s += l[k] * w[k * HID + c];
  const float* e = emb + row * HID;
  for (int k = 0; k < HID; ++k)   s += e[k] * w[(LATD + k) * HID + c];
  const float* qq = qe + row * QENCD;
  for (int k = 0; k < QENCD; ++k) s += qq[k] * w[(LATD + HID + k) * HID + c];
  y[tid] = fmaxf(s, 0.0f);
}

__global__ void k_nh2(const float* __restrict__ hh, const float* __restrict__ w2,
                      const float* __restrict__ b2, float* __restrict__ outp) {
  int tid = blockIdx.x * 256 + threadIdx.x;           // 1024
  const float* xr = hh + tid * HID;
  float s = b2[0];
  for (int k = 0; k < HID; ++k) s += xr[k] * w2[k];
  outp[tid] = s;
}

// ---------------- weight prep ----------------
// wfrag: f32 [384,256] -> bf16 WMMA B-fragment layout.
// fragment (l, c, t): 32 lanes x 16 bf16. lane<16: K = c*32 + {0..7,16..23}; lane>=16: +8.
// N = t*16 + lane%16.  Stored at wfrag[((l*12 + c)*16 + t)*512 + lane*16 + v].

__global__ void k_prep(const float* __restrict__ ew1, const float* __restrict__ ow1,
                       unsigned short* __restrict__ wfrag) {
  int tid = blockIdx.x * 256 + threadIdx.x;           // 196608
  int v    = tid & 15;
  int lane = (tid >> 4) & 31;
  int frag = tid >> 9;
  int t = frag & 15;
  int x = frag >> 4;          // 0..23
  int c = x % 12;
  int l = x / 12;
  int hi = lane >> 4;
  int klocal = (v < 8 ? v : v + 8) + (hi << 3);
  int K = c * 32 + klocal;
  int N = (t << 4) + (lane & 15);
  const float* W = l ? ow1 : ew1;
  wfrag[tid] = f32_to_bf16(W[K * HID + N]);
}

// transpose oh_w2 [256,8] -> [8][256] so layer-2 reads are contiguous / vectorizable
__global__ void k_prep2(const float* __restrict__ w2, float* __restrict__ w2t) {
  int tid = blockIdx.x * 256 + threadIdx.x;           // 2048
  int n = tid >> 3, o = tid & 7;
  w2t[o * HID + n] = w2[n * NOPSD + o];
}

// ---------------- dominant kernel: all-pairs MLPs via v_wmma_f32_16x16x32_bf16 ----------
// 2 M-tiles (32 pairs, same i) per workgroup: each B fragment feeds 2 WMMAs, and the
// lat_i A-chunks (c<4) are shared between both M-tiles.

__global__ __launch_bounds__(128) void k_pair(
    const unsigned short* __restrict__ latb,
    const unsigned short* __restrict__ wfrag,
    const float* __restrict__ eh_b1, const float* __restrict__ eh_w2, const float* __restrict__ eh_b2,
    const float* __restrict__ oh_b1, const float* __restrict__ oh_w2t, const float* __restrict__ oh_b2,
    float* __restrict__ out) {
  __shared__ float hbuf[2][32][HID];                  // 64 KB

  const int g   = blockIdx.x;                         // 8192 tiles of 32 pairs
  const int e0  = g << 5;
  const int b   = e0 >> 16;
  const int rem = e0 & 65535;
  const int i   = rem >> 8;
  const int j0  = rem & 255;

  const int tid  = threadIdx.x;
  const int lane = tid & 31;
  const int wave = tid >> 5;
  const int m    = lane & 15;
  const int hi   = lane >> 4;

  // dword-granularity row pointers (64 dwords = 128 bf16 per lat row)
  const unsigned int* rowi  = (const unsigned int*)(latb + (((size_t)b * NNODE + i) << 7));
  const unsigned int* rowj0 = (const unsigned int*)(latb + (((size_t)b * NNODE + (j0 + m)) << 7));
  const unsigned int* rowj1 = rowj0 + 64;             // j0 + m + 1

  const v8f vz = {0.f, 0.f, 0.f, 0.f, 0.f, 0.f, 0.f, 0.f};
  v8f accE[2][4] = {{vz, vz, vz, vz}, {vz, vz, vz, vz}};
  v8f accO[2][4] = {{vz, vz, vz, vz}, {vz, vz, vz, vz}};

  #pragma unroll
  for (int c = 0; c < 12; ++c) {
    // build A fragments for both M-tiles, K chunk c (32 wide).
    // lane run offsets in dwords: base..base+3 and base+8..base+11
    const int base = ((c & 3) << 4) + (hi << 2);
    v8i au0, au1;
    if (c < 4) {
      v4i x0 = *(const v4i*)(rowi + base);
      v4i x1 = *(const v4i*)(rowi + base + 8);
      au0 = __builtin_shufflevector(x0, x1, 0, 1, 2, 3, 4, 5, 6, 7);
      au1 = au0;                                      // lat_i chunk identical for both M-tiles
    } else if (c < 8) {
      v4i y0 = *(const v4i*)(rowj0 + base);
      v4i y1 = *(const v4i*)(rowj0 + base + 8);
      v4i z0 = *(const v4i*)(rowj1 + base);
      v4i z1 = *(const v4i*)(rowj1 + base + 8);
      au0 = __builtin_shufflevector(y0, y1, 0, 1, 2, 3, 4, 5, 6, 7);
      au1 = __builtin_shufflevector(z0, z1, 0, 1, 2, 3, 4, 5, 6, 7);
    } else {
      v4i x0 = *(const v4i*)(rowi + base);
      v4i x1 = *(const v4i*)(rowi + base + 8);
      v4i y0 = *(const v4i*)(rowj0 + base);
      v4i y1 = *(const v4i*)(rowj0 + base + 8);
      v4i z0 = *(const v4i*)(rowj1 + base);
      v4i z1 = *(const v4i*)(rowj1 + base + 8);
      v4i d0, d1, e0v, e1v;
      #pragma unroll
      for (int k = 0; k < 4; ++k) {
        d0[k]  = pk_absdiff_bf16(x0[k], y0[k]);
        d1[k]  = pk_absdiff_bf16(x1[k], y1[k]);
        e0v[k] = pk_absdiff_bf16(x0[k], z0[k]);
        e1v[k] = pk_absdiff_bf16(x1[k], z1[k]);
      }
      au0 = __builtin_shufflevector(d0, d1, 0, 1, 2, 3, 4, 5, 6, 7);
      au1 = __builtin_shufflevector(e0v, e1v, 0, 1, 2, 3, 4, 5, 6, 7);
    }
    v16bf aA = __builtin_bit_cast(v16bf, au0);
    v16bf aB = __builtin_bit_cast(v16bf, au1);

    #pragma unroll
    for (int t = 0; t < 4; ++t) {
      const int nt = (wave << 2) + t;
      const unsigned short* fE = wfrag + (((size_t)((c << 4) + nt)) << 9) + (lane << 4);
      const unsigned short* fO = wfrag + (((size_t)(((12 + c) << 4) + nt)) << 9) + (lane << 4);
      v16bf be = __builtin_bit_cast(v16bf, *(const v8i*)fE);
      v16bf bo = __builtin_bit_cast(v16bf, *(const v8i*)fO);
      accE[0][t] = WMMA_BF16(aA, be, accE[0][t]);
      accE[1][t] = WMMA_BF16(aB, be, accE[1][t]);
      accO[0][t] = WMMA_BF16(aA, bo, accO[0][t]);
      accO[1][t] = WMMA_BF16(aB, bo, accO[1][t]);
    }
  }

  // bias + ReLU, scatter C fragments to LDS (lane: N = nt*16 + lane%16, rows r + 8*hi + 16*mt)
  #pragma unroll
  for (int t = 0; t < 4; ++t) {
    const int n = (((wave << 2) + t) << 4) + (lane & 15);
    const float be1 = eh_b1[n];
    const float bo1 = oh_b1[n];
    #pragma unroll
    for (int mt = 0; mt < 2; ++mt) {
      #pragma unroll
      for (int r = 0; r < 8; ++r) {
        const int mm = r + (hi << 3) + (mt << 4);
        hbuf[0][mm][n] = fmaxf(accE[mt][t][r] + be1, 0.0f);
        hbuf[1][mm][n] = fmaxf(accO[mt][t][r] + bo1, 0.0f);
      }
    }
  }
  __syncthreads();

  // layer 2 (tiny), float4-vectorized: op logits, 256 outputs (32 rows x 8 ops), 2 per thread
  #pragma unroll
  for (int rep = 0; rep < 2; ++rep) {
    const int idx = tid + (rep << 7);
    const int mm = idx >> 3;
    const int o  = idx & 7;
    const float4* h4 = (const float4*)&hbuf[1][mm][0];
    const float4* w4 = (const float4*)(oh_w2t + (o << 8));
    float s = 0.0f;
    #pragma unroll 8
    for (int n = 0; n < HID / 4; ++n) {
      float4 h = h4[n];
      float4 w = w4[n];
      s += h.x * w.x + h.y * w.y + h.z * w.z + h.w * w.w;
    }
    s += oh_b2[o];
    out[OP_OFF + ((((size_t)b * NNODE + i) * NNODE + (j0 + mm)) << 3) + o] = s;
  }
  // edge logits, 32 outputs, diagonal mask applied at store
  if (tid < 32) {
    const int mm = tid;
    const float4* h4 = (const float4*)&hbuf[0][mm][0];
    const float4* w4 = (const float4*)eh_w2;
    float s = 0.0f;
    #pragma unroll 8
    for (int n = 0; n < HID / 4; ++n) {
      float4 h = h4[n];
      float4 w = w4[n];
      s += h.x * w.x + h.y * w.y + h.z * w.z + h.w * w.w;
    }
    s += eh_b2[0];
    const int jj = j0 + mm;
    out[EDGE_OFF + ((size_t)b * NNODE + i) * NNODE + jj] = (jj == i) ? -8.0f : s;
  }
}

// ---------------- launcher ----------------

extern "C" void kernel_launch(void* const* d_in, const int* in_sizes, int n_in,
                              void* d_out, int out_size, void* d_ws, size_t ws_size,
                              hipStream_t stream) {
  const float* obs    = (const float*)d_in[0];
  const float* latent = (const float*)d_in[1];
  const float* query  = (const float*)d_in[2];
  const float* enc_w1 = (const float*)d_in[3];
  const float* enc_b1 = (const float*)d_in[4];
  const float* enc_w2 = (const float*)d_in[5];
  const float* enc_b2 = (const float*)d_in[6];
  const float* upd_w  = (const float*)d_in[7];
  const float* upd_b  = (const float*)d_in[8];
  const float* gru_wi = (const float*)d_in[9];
  const float* gru_bi = (const float*)d_in[10];
  const float* gru_wh = (const float*)d_in[11];
  const float* gru_bh = (const float*)d_in[12];
  const float* eh_w1  = (const float*)d_in[13];
  const float* eh_b1  = (const float*)d_in[14];
  const float* eh_w2  = (const float*)d_in[15];
  const float* eh_b2  = (const float*)d_in[16];
  const float* oh_w1  = (const float*)d_in[17];
  const float* oh_b1  = (const float*)d_in[18];
  const float* oh_w2  = (const float*)d_in[19];
  const float* oh_b2  = (const float*)d_in[20];
  const float* q_w    = (const float*)d_in[21];
  const float* q_b    = (const float*)d_in[22];
  const float* nh_w1  = (const float*)d_in[23];
  const float* nh_b1  = (const float*)d_in[24];
  const float* nh_w2  = (const float*)d_in[25];
  const float* nh_b2  = (const float*)d_in[26];
  float* out = (float*)d_out;

  // workspace carve-up (f32 region, then bf16/ushort region, then transposed w2); ~5.2 MB
  float* h1   = (float*)d_ws;                    // 1024*256
  float* emb  = h1   + 262144;                   // 1024*256
  float* updv = emb  + 262144;                   // 1024*128
  float* latf = updv + 131072;                   // 1024*128
  float* qe   = latf + 131072;                   // 1024*64
  float* nhh  = qe   + 65536;                    // 1024*256
  unsigned short* latb  = (unsigned short*)(nhh + 262144);   // 1024*128 bf16
  unsigned short* wfrag = latb + 131072;                     // 2*12*16*512 bf16
  float* ohw2t = (float*)(wfrag + 196608);                   // 8*256 f32 (16B aligned)

  k_prep <<<768, 256, 0, stream>>>(eh_w1, oh_w1, wfrag);
  k_prep2<<<8, 256, 0, stream>>>(oh_w2, ohw2t);
  k_enc1 <<<1024, 256, 0, stream>>>(obs, enc_w1, enc_b1, h1);
  k_enc2 <<<1024, 256, 0, stream>>>(h1, enc_w2, enc_b2, emb);
  k_upd  <<<512, 256, 0, stream>>>(emb, upd_w, upd_b, updv);
  k_gru  <<<512, 256, 0, stream>>>(updv, latent, gru_wi, gru_bi, gru_wh, gru_bh, latf, latb);
  k_qemb <<<256, 256, 0, stream>>>(query, q_w, q_b, qe);
  k_nh1  <<<1024, 256, 0, stream>>>(latf, emb, qe, nh_w1, nh_b1, nhh);
  k_nh2  <<<4, 256, 0, stream>>>(nhh, nh_w2, nh_b2, out + NEXT_OFF);
  k_pair <<<8192, 128, 0, stream>>>(latb, wfrag,
                                    eh_b1, eh_w2, eh_b2,
                                    oh_b1, ohw2t, oh_b2, out);
}
